// ResNetBlock_8366596293052
// MI455X (gfx1250) — matrix operand order repaired and checked by reading, not compiled, not run on hardware
//
#include <hip/hip_runtime.h>

// ---------------------------------------------------------------------------
// Sparse ResNet block (MinkowskiEngine-style) for MI455X / gfx1250.
//   out = conv2(relu(bn2(conv1(relu(bn1(x)))))) + (x @ Wres + bres)
// Core GEMMs: v_wmma_f32_16x16x32_f16 (f16 in, f32 accumulate).
// Weights double-buffered in LDS via async global->LDS copies (ASYNCcnt).
// ---------------------------------------------------------------------------

typedef __attribute__((ext_vector_type(16))) _Float16     v16h;
typedef __attribute__((ext_vector_type(8)))  float        v8f;
typedef __attribute__((ext_vector_type(4)))  float        v4f;
typedef __attribute__((ext_vector_type(4)))  unsigned int v4u;
typedef __attribute__((ext_vector_type(2)))  unsigned int v2u;
typedef __attribute__((__vector_size__(16))) int          b128_t;  // builtin's param type

union Frag16 {            // one wave-striped 16x32 (A) or 32x16 (B) f16 fragment
  v16h v;                 // 8 VGPRs
  v4u  u[2];              // two 16-byte chunks (b128 loads)
};

#define BN_EPS 1e-5f

// gfx1250 async global->LDS path (ASYNCcnt); guarded so either toolchain builds.
#if defined(__HIP_DEVICE_COMPILE__) &&                                   \
    __has_builtin(__builtin_amdgcn_global_load_async_to_lds_b128) &&     \
    __has_builtin(__builtin_amdgcn_s_wait_asynccnt)
#define USE_ASYNC_LDS 1
#else
#define USE_ASYNC_LDS 0
#endif

__device__ __forceinline__ void async_copy_b128(const _Float16* src, _Float16* lds_dst) {
#if USE_ASYNC_LDS
  // ROCm clang declares the builtin with generic int4* params; the backend
  // infers global/LDS address spaces from the pointer values.
  __builtin_amdgcn_global_load_async_to_lds_b128(
      (b128_t*)(unsigned long long)src,   // global source (const dropped via int round-trip)
      (b128_t*)lds_dst,                   // LDS destination (flat pointer into __shared__)
      0, 0);
#else
  *(v4u*)lds_dst = *(const v4u*)src;
#endif
}

__device__ __forceinline__ void stage_wait() {
#if USE_ASYNC_LDS
  __builtin_amdgcn_s_wait_asynccnt(0);
#endif
  __syncthreads();
}

// ---------------------------------------------------------------------------
// prep: zero stat accumulators + sentinel rows; convert & transpose weights
//   W1 [27][32][64] f32 -> Wt1 [27][64][32] f16   (cout-major => contiguous K)
//   W2 [27][64][64] f32 -> Wt2 [27][64][64] f16
// ---------------------------------------------------------------------------
__global__ __launch_bounds__(256) void prep_kernel(
    const float* __restrict__ W1, const float* __restrict__ W2,
    _Float16* __restrict__ Wt1, _Float16* __restrict__ Wt2,
    float* __restrict__ s1, float* __restrict__ s2,
    _Float16* __restrict__ y0, _Float16* __restrict__ y1, int Nv) {
  const int idx = blockIdx.x * blockDim.x + threadIdx.x;
  if (idx < 27 * 32 * 64) {
    const int k = idx / (32 * 64), rem = idx % (32 * 64);
    const int c = rem / 64, o = rem % 64;
    Wt1[((size_t)k * 64 + o) * 32 + c] = (_Float16)W1[idx];
  }
  if (idx < 27 * 64 * 64) {
    const int k = idx / (64 * 64), rem = idx % (64 * 64);
    const int c = rem / 64, o = rem % 64;
    Wt2[((size_t)k * 64 + o) * 64 + c] = (_Float16)W2[idx];
  }
  if (idx < 64)  s1[idx] = 0.0f;
  if (idx < 128) s2[idx] = 0.0f;
  if (idx < 32)  y0[(long long)Nv * 32 + idx] = (_Float16)0.0f;  // zero sentinel row
  if (idx < 64)  y1[(long long)Nv * 64 + idx] = (_Float16)0.0f;
}

// ---------------------------------------------------------------------------
// stats over x [N,32]: s[0..31]=sum, s[32..63]=sumsq  (per channel)
// ---------------------------------------------------------------------------
__global__ __launch_bounds__(256) void stats32_kernel(
    const float* __restrict__ x, float* __restrict__ s, int Nv) {
  __shared__ float sh[64];
  const int tid = threadIdx.x;
  if (tid < 64) sh[tid] = 0.0f;
  __syncthreads();
  const int cg = (tid & 7) * 4;
  float s0 = 0, s1_ = 0, s2_ = 0, s3 = 0, q0 = 0, q1 = 0, q2 = 0, q3 = 0;
  const long long total4 = (long long)Nv * 8;               // float4s
  const long long stride = (long long)gridDim.x * blockDim.x;
  for (long long f = (long long)blockIdx.x * blockDim.x + tid; f < total4; f += stride) {
    v4f v = ((const v4f*)x)[f];
    s0 += v.x; s1_ += v.y; s2_ += v.z; s3 += v.w;
    q0 += v.x * v.x; q1 += v.y * v.y; q2 += v.z * v.z; q3 += v.w * v.w;
  }
  atomicAdd(&sh[cg + 0], s0); atomicAdd(&sh[cg + 1], s1_);
  atomicAdd(&sh[cg + 2], s2_); atomicAdd(&sh[cg + 3], s3);
  atomicAdd(&sh[32 + cg + 0], q0); atomicAdd(&sh[32 + cg + 1], q1);
  atomicAdd(&sh[32 + cg + 2], q2); atomicAdd(&sh[32 + cg + 3], q3);
  __syncthreads();
  if (tid < 64) atomicAdd(&s[tid], sh[tid]);
}

// ---------------------------------------------------------------------------
// bn1 + relu -> f16 y0  (one row/thread), and exact f32 residual -> out
// ---------------------------------------------------------------------------
__global__ __launch_bounds__(256) void bn1_residual_kernel(
    const float* __restrict__ x, const float* __restrict__ s1,
    const float* __restrict__ g1, const float* __restrict__ b1,
    const float* __restrict__ Wres, const float* __restrict__ bres,
    _Float16* __restrict__ y0, float* __restrict__ out, int Nv) {
  __shared__ __align__(16) float wsh[32 * 64];
  __shared__ __align__(16) float bsh[64];
  __shared__ float scl[32], sft[32];
  const int tid = threadIdx.x;
  for (int i = tid; i < 2048; i += 256) wsh[i] = Wres[i];
  if (tid < 64) bsh[tid] = bres[tid];
  if (tid < 32) {
    const float inv = 1.0f / (float)Nv;
    const float m = s1[tid] * inv;
    const float var = s1[32 + tid] * inv - m * m;
    const float sc = g1[tid] * rsqrtf(var + BN_EPS);
    scl[tid] = sc;
    sft[tid] = b1[tid] - m * sc;
  }
  __syncthreads();
  const long long r = (long long)blockIdx.x * 256 + tid;
  if (r >= Nv) return;
  float xi[32];
  const v4f* xr = (const v4f*)(x + r * 32);
#pragma unroll
  for (int j = 0; j < 8; ++j) {
    v4f t = xr[j];
    xi[4 * j + 0] = t.x; xi[4 * j + 1] = t.y; xi[4 * j + 2] = t.z; xi[4 * j + 3] = t.w;
  }
  union { _Float16 h[32]; v4u u[4]; } yo;
#pragma unroll
  for (int c = 0; c < 32; ++c)
    yo.h[c] = (_Float16)fmaxf(fmaf(xi[c], scl[c], sft[c]), 0.0f);
  v4u* py = (v4u*)(y0 + r * 32);
#pragma unroll
  for (int j = 0; j < 4; ++j) py[j] = yo.u[j];
  // residual = x @ Wres + bres (full f32; LDS reads are lane-uniform broadcasts)
  v4f acc[16];
  const v4f* bv = (const v4f*)bsh;
#pragma unroll
  for (int j = 0; j < 16; ++j) acc[j] = bv[j];
  for (int c = 0; c < 32; ++c) {
    const float xv = xi[c];
    const v4f* wrow = (const v4f*)(wsh + c * 64);
#pragma unroll
    for (int j = 0; j < 16; ++j) {
      v4f w = wrow[j];
      acc[j].x = fmaf(xv, w.x, acc[j].x);
      acc[j].y = fmaf(xv, w.y, acc[j].y);
      acc[j].z = fmaf(xv, w.z, acc[j].z);
      acc[j].w = fmaf(xv, w.w, acc[j].w);
    }
  }
  v4f* po = (v4f*)(out + r * 64);
#pragma unroll
  for (int j = 0; j < 16; ++j) po[j] = acc[j];
}

// ---------------------------------------------------------------------------
// Gather-GEMM sparse conv on WMMA f16 with LDS-staged, double-buffered weights:
//   out[i,:] (+)= sum_k  yin[nbr[i,k], :] @ W[k]       (Cout = 64)
//
// Per wave: 2 M-tiles (32 rows) x 4 N-tiles (full Cout); K loop over 27*CIN.
// Per k, the 64 x CIN weight slab lives in LDS (pitch CIN+8 f16 -> each lane's
// b128 starts on a distinct bank group => conflict-free ds_load_b128), staged
// one iteration ahead by async global->LDS copies overlapped with WMMA work.
// A fragment (16x32 f16): lane m=l%16; lanes<16 hold K 0-15, lanes>=16
// hold K 16-31 -> two b128 gathers from row-major f16 rows.
// B fragment (32x16 f16): lane col n=l%16; lanes<16 K 0-15, lanes>=16 K 16-31.
// Optionally fuses BN2 channel sums/sumsq (sentinel rows contribute exact 0).
// All waves stay in the barrier protocol; EXEC is full around every WMMA.
// ---------------------------------------------------------------------------
template <int CIN, bool ADD_OUT, bool STATS>
__global__ __launch_bounds__(256) void sparse_conv_wmma(
    const _Float16* __restrict__ yin,   // (N+1) x CIN
    const long long* __restrict__ nbr,  // N x 27 (value N = sentinel)
    const _Float16* __restrict__ Wt,    // 27 x 64 x CIN
    float* __restrict__ out,            // N x 64
    float* __restrict__ stats,          // [128] sum/sumsq (if STATS)
    int Nv) {
  constexpr int KT = CIN / 32;
  constexpr int P = CIN + 8;            // padded LDS pitch (f16), 16B-aligned rows
  constexpr int CPR = CIN / 8;          // b128 chunks per cout-row
  __shared__ __align__(16) _Float16 ldsW[2][64 * P];
  __shared__ float sh[128];
  const int tid = threadIdx.x;
  if (STATS && tid < 128) sh[tid] = 0.0f;
  const int lane = tid & 31;
  const int half = lane >> 4;           // which 16-lane half
  const int l16 = lane & 15;
  const int wave = tid >> 5;
  const long long mrow0 = ((long long)blockIdx.x * 8 + wave) * 32;

  // stage slab k=0 into buffer 0
  for (int c = tid; c < 64 * CPR; c += 256)
    async_copy_b128(Wt + (c / CPR) * CIN + (c % CPR) * 8,
                    &ldsW[0][(c / CPR) * P + (c % CPR) * 8]);
  stage_wait();                         // also publishes sh[] zeros

  v8f zero = {};
  v8f acc[2][4];
#pragma unroll
  for (int mt = 0; mt < 2; ++mt)
#pragma unroll
    for (int nt = 0; nt < 4; ++nt) acc[mt][nt] = zero;
  float lsum[4] = {0.f, 0.f, 0.f, 0.f}, lsq[4] = {0.f, 0.f, 0.f, 0.f};

  for (int k = 0; k < 27; ++k) {
    // prefetch next slab into the other buffer (overlaps with compute below)
    if (k + 1 < 27) {
      const _Float16* wn = Wt + (size_t)(k + 1) * 64 * CIN;
      _Float16* dn = ldsW[(k + 1) & 1];
      for (int c = tid; c < 64 * CPR; c += 256)
        async_copy_b128(wn + (c / CPR) * CIN + (c % CPR) * 8,
                        &dn[(c / CPR) * P + (c % CPR) * 8]);
    }
    // ---- B fragments from LDS (conflict-free ds_load_b128) ----
    const _Float16* cur = ldsW[k & 1];
    Frag16 bfrag[KT][4];
#pragma unroll
    for (int nt = 0; nt < 4; ++nt) {
#pragma unroll
      for (int kt = 0; kt < KT; ++kt) {
        const _Float16* bp = cur + (nt * 16 + l16) * P + kt * 32 + half * 16;
        bfrag[kt][nt].u[0] = *(const v4u*)(bp);
        bfrag[kt][nt].u[1] = *(const v4u*)(bp + 8);
      }
    }
#pragma unroll
    for (int mt = 0; mt < 2; ++mt) {
      // ---- gather A fragment rows via kernel map (sentinel row => zeros) ----
      const long long r = mrow0 + mt * 16 + l16;
      const long long idx = (r < Nv) ? nbr[r * 27 + k] : (long long)Nv;
      const _Float16* arow = yin + idx * CIN;
      Frag16 afrag[KT];
#pragma unroll
      for (int kt = 0; kt < KT; ++kt) {
        const _Float16* ap = arow + kt * 32 + half * 16;
        afrag[kt].u[0] = *(const v4u*)(ap);
        afrag[kt].u[1] = *(const v4u*)(ap + 8);
      }
#pragma unroll
      for (int kt = 0; kt < KT; ++kt)
#pragma unroll
        for (int nt = 0; nt < 4; ++nt)
          acc[mt][nt] = __builtin_amdgcn_wmma_f32_16x16x32_f16(
              false, afrag[kt].v, false, bfrag[kt][nt].v,
              (short)0, acc[mt][nt], false, false);
    }
    stage_wait();   // next slab resident & visible to the whole workgroup
  }

  // ---- write back; C/D layout: VGPR v -> row m = v + half*8, col = l16 ----
#pragma unroll
  for (int mt = 0; mt < 2; ++mt) {
#pragma unroll
    for (int nt = 0; nt < 4; ++nt) {
      const int ch = nt * 16 + l16;
#pragma unroll
      for (int v = 0; v < 8; ++v) {
        const long long row = mrow0 + mt * 16 + half * 8 + v;
        const float val = acc[mt][nt][v];
        if (row < Nv) {
          float* po = out + row * 64 + ch;
          if (ADD_OUT) *po = *po + val; else *po = val;
        }
        if (STATS) { lsum[nt] += val; lsq[nt] += val * val; }
      }
    }
  }
  if (STATS) {
#pragma unroll
    for (int nt = 0; nt < 4; ++nt) {
      const int ch = nt * 16 + l16;
      atomicAdd(&sh[ch], lsum[nt]);
      atomicAdd(&sh[64 + ch], lsq[nt]);
    }
    __syncthreads();
    if (tid < 128) atomicAdd(&stats[tid], sh[tid]);
  }
}

// ---------------------------------------------------------------------------
// bn2 + relu -> f16 y1 (one row/thread)
// ---------------------------------------------------------------------------
__global__ __launch_bounds__(256) void bn2_kernel(
    const float* __restrict__ z1, const float* __restrict__ s2,
    const float* __restrict__ g2, const float* __restrict__ b2,
    _Float16* __restrict__ y1, int Nv) {
  __shared__ float scl[64], sft[64];
  const int tid = threadIdx.x;
  if (tid < 64) {
    const float inv = 1.0f / (float)Nv;
    const float m = s2[tid] * inv;
    const float var = s2[64 + tid] * inv - m * m;
    const float sc = g2[tid] * rsqrtf(var + BN_EPS);
    scl[tid] = sc;
    sft[tid] = b2[tid] - m * sc;
  }
  __syncthreads();
  const long long r = (long long)blockIdx.x * 256 + tid;
  if (r >= Nv) return;
  const v4f* zr = (const v4f*)(z1 + r * 64);
#pragma unroll
  for (int j = 0; j < 16; ++j) {
    v4f v = zr[j];
    union { _Float16 h[4]; v2u u; } t;
    t.h[0] = (_Float16)fmaxf(fmaf(v.x, scl[4 * j + 0], sft[4 * j + 0]), 0.0f);
    t.h[1] = (_Float16)fmaxf(fmaf(v.y, scl[4 * j + 1], sft[4 * j + 1]), 0.0f);
    t.h[2] = (_Float16)fmaxf(fmaf(v.z, scl[4 * j + 2], sft[4 * j + 2]), 0.0f);
    t.h[3] = (_Float16)fmaxf(fmaf(v.w, scl[4 * j + 3], sft[4 * j + 3]), 0.0f);
    *(v2u*)(y1 + r * 64 + 4 * j) = t.u;
  }
}

// ---------------------------------------------------------------------------
// launch: x, nbr, W1, W2, Wres, bres, g1, b1, g2, b2
// ---------------------------------------------------------------------------
extern "C" void kernel_launch(void* const* d_in, const int* in_sizes, int n_in,
                              void* d_out, int out_size, void* d_ws, size_t ws_size,
                              hipStream_t stream) {
  (void)n_in; (void)out_size; (void)ws_size;
  const float*     x    = (const float*)d_in[0];
  const long long* nbr  = (const long long*)d_in[1];
  const float*     W1   = (const float*)d_in[2];
  const float*     W2   = (const float*)d_in[3];
  const float*     Wres = (const float*)d_in[4];
  const float*     bres = (const float*)d_in[5];
  const float*     g1   = (const float*)d_in[6];
  const float*     b1   = (const float*)d_in[7];
  const float*     g2   = (const float*)d_in[8];
  const float*     b2   = (const float*)d_in[9];
  float* out = (float*)d_out;
  const int N = in_sizes[0] / 32;

  // ---- carve workspace (~224 MB) ----
  char* base = (char*)d_ws;
  size_t off = 0;
  auto carve = [&](size_t bytes) -> void* {
    off = (off + 255) & ~(size_t)255;
    void* r = base + off;
    off += bytes;
    return r;
  };
  _Float16* y0  = (_Float16*)carve((size_t)(N + 1) * 32 * sizeof(_Float16));
  _Float16* y1  = (_Float16*)carve((size_t)(N + 1) * 64 * sizeof(_Float16));
  float*    z1  = (float*)   carve((size_t)N * 64 * sizeof(float));
  _Float16* Wt1 = (_Float16*)carve((size_t)27 * 64 * 32 * sizeof(_Float16));
  _Float16* Wt2 = (_Float16*)carve((size_t)27 * 64 * 64 * sizeof(_Float16));
  float*    s1  = (float*)   carve(64 * sizeof(float));
  float*    s2  = (float*)   carve(128 * sizeof(float));

  const int rowBlocks  = (N + 255) / 256;
  const int convBlocks = ((N + 31) / 32 + 7) / 8;   // 8 waves/block, 32 rows/wave

  prep_kernel<<<(27 * 64 * 64 + 255) / 256, 256, 0, stream>>>(
      W1, W2, Wt1, Wt2, s1, s2, y0, y1, N);
  stats32_kernel<<<1024, 256, 0, stream>>>(x, s1, N);
  bn1_residual_kernel<<<rowBlocks, 256, 0, stream>>>(
      x, s1, g1, b1, Wres, bres, y0, out, N);
  sparse_conv_wmma<32, false, true><<<convBlocks, 256, 0, stream>>>(
      y0, nbr, Wt1, z1, s2, N);
  bn2_kernel<<<rowBlocks, 256, 0, stream>>>(z1, s2, g2, b2, y1, N);
  sparse_conv_wmma<64, true, false><<<convBlocks, 256, 0, stream>>>(
      y1, nbr, Wt2, out, (float*)nullptr, N);
}
